// MHAttention_26654567039351
// MI455X (gfx1250) — matrix operand order repaired and checked by reading, not compiled, not run on hardware
//
#include <hip/hip_runtime.h>
#include <hip/hip_bf16.h>

// MHA forward (B=4, S=2048, H=12, hd=64, hidden=768; fp32 in/out).
// Pass 1: QKV projection via v_wmma_f32_16x16x32_f16 -> f16 workspace.
//         Q,K in [B*H][S][64]; V stored TRANSPOSED [B*H][64][S] so the PV
//         B-fragments are contiguous in LDS.
// Pass 2: flash-attention; K/V tiles staged by the Tensor Data Mover
//         (tensor_load_to_lds, TENSORcnt) into double-buffered LDS; QK^T and
//         PV via WMMA; online softmax in f32.

typedef _Float16 v16h __attribute__((ext_vector_type(16)));
typedef float    v8f  __attribute__((ext_vector_type(8)));
typedef unsigned int u32x4 __attribute__((ext_vector_type(4)));
typedef int          i32x8 __attribute__((ext_vector_type(8)));
typedef int          i32x4 __attribute__((ext_vector_type(4)));

constexpr int kB  = 4;
constexpr int kS  = 2048;
constexpr int kD  = 768;
constexpr int kH  = 12;
constexpr int kHD = 64;
constexpr float kNegBig = -1.0e31f;

// A-matrix (16x32 f16) K index for fragment element j (ISA 7.12.2).
__device__ __forceinline__ int kmapA(int lane, int j) {
  return ((j & 8) << 1) + ((lane >> 4) << 3) + (j & 7);
}
// B-matrix (32x16 f16) K index: same K order as the A fragment.
__device__ __forceinline__ int kmapB(int lane, int j) {
  return ((j & 8) << 1) + ((lane >> 4) << 3) + (j & 7);
}

// Issue a TDM 2D tile load (f16 elements) into LDS. D# per ISA ch.8:
//  group0: count=1 | lds_addr | global_addr[56:0] | type=2
//  group1: data_size=1(2B), tensor_dim0/1, tile_dim0/1, tensor_dim0_stride
__device__ __forceinline__ void tdm_load_tile_2d(
    unsigned lds_off, const void* gptr,
    unsigned tensor_d0, unsigned tensor_d1,
    unsigned tile_d0, unsigned tile_d1, unsigned stride0_elems)
{
  unsigned long long ga = (unsigned long long)gptr;
  u32x4 g0 = { 1u,                                   // count=1 (valid D#)
               lds_off,                              // lds_addr (bytes)
               (unsigned)ga,                         // global_addr[31:0]
               ((unsigned)(ga >> 32) & 0x01FFFFFFu) | (2u << 30) }; // [56:32]|type=2
  i32x8 g1;
  g1[0] = (int)(1u << 16);                           // data_size=1 -> 2 bytes
  g1[1] = (int)((tensor_d0 & 0xFFFFu) << 16);        // tensor_dim0[15:0]
  g1[2] = (int)((tensor_d0 >> 16) | ((tensor_d1 & 0xFFFFu) << 16));
  g1[3] = (int)((tensor_d1 >> 16) | (tile_d0 << 16)); // tensor_dim1[31:16]|tile_dim0
  g1[4] = (int)(tile_d1 & 0xFFFFu);                  // tile_dim1 (tile_dim2=0)
  g1[5] = (int)stride0_elems;                        // tensor_dim0_stride[31:0]
  g1[6] = 0;
  g1[7] = 0;
  i32x4 z = {0, 0, 0, 0};
#if defined(__clang_major__) && (__clang_major__ >= 23)
  i32x8 z8 = {0, 0, 0, 0, 0, 0, 0, 0};
  __builtin_amdgcn_tensor_load_to_lds(g0, g1, z, z, z8, 0);
#else
  __builtin_amdgcn_tensor_load_to_lds(g0, g1, z, z, 0);
#endif
}

__global__ __launch_bounds__(256) void qkv_proj_kernel(
    const float* __restrict__ toks,
    const float* __restrict__ Wq, const float* __restrict__ bq,
    const float* __restrict__ Wk, const float* __restrict__ bk,
    const float* __restrict__ Wv, const float* __restrict__ bv,
    _Float16* __restrict__ Qh, _Float16* __restrict__ Kh,
    _Float16* __restrict__ Vh)
{
  const int lane = threadIdx.x & 31;
  const int wave = threadIdx.x >> 5;             // 0..7
  const int mat  = blockIdx.y / 6;               // 0=Q 1=K 2=V
  const int ct   = (blockIdx.y % 6) * 8 + wave;  // col tile 0..47
  const int n    = lane & 15;
  const int col  = ct * 16 + n;
  const int m0   = blockIdx.x * 16;              // row tile base (rows = b*S+s)

  const float* W; const float* bias; _Float16* dst;
  if (mat == 0)      { W = Wq; bias = bq; dst = Qh; }
  else if (mat == 1) { W = Wk; bias = bk; dst = Kh; }
  else               { W = Wv; bias = bv; dst = Vh; }

  const int arow = m0 + n;
  const float bcol = bias[col];

  v8f c = {};
  for (int k0 = 0; k0 < kD; k0 += 32) {
    v16h a, bf;
#pragma unroll
    for (int j = 0; j < 16; ++j) {
      const int ka = k0 + kmapA(lane, j);
      const int kb = k0 + kmapB(lane, j);
      a[j]  = (_Float16)toks[(size_t)arow * kD + ka];
      bf[j] = (_Float16)W[(size_t)kb * kD + col];
    }
    c = __builtin_amdgcn_wmma_f32_16x16x32_f16(false, a, false, bf,
                                               (short)0, c, false, false);
  }

  const int mbase = (lane >> 4) << 3;
  const int hh = col >> 6;
  const int dd = col & 63;
#pragma unroll
  for (int j = 0; j < 8; ++j) {
    const int r  = m0 + mbase + j;               // global row = b*S + s
    const int bb = r >> 11;
    const int ss = r & (kS - 1);
    const size_t bh = (size_t)bb * kH + hh;
    const _Float16 val = (_Float16)(c[j] + bcol);
    if (mat == 2)  // V transposed: [bh][d][s]
      dst[(bh * kHD + dd) * kS + ss] = val;
    else           // Q,K: [bh][s][d]
      dst[(bh * kS + ss) * kHD + dd] = val;
  }
}

__global__ __launch_bounds__(128) void attn_kernel(
    const _Float16* __restrict__ Qh, const _Float16* __restrict__ Kh,
    const _Float16* __restrict__ Vh, const int* __restrict__ masks,
    float* __restrict__ out)
{
  __shared__ _Float16 Kbuf[2][32 * 64];   // [key][d]       8 KB
  __shared__ _Float16 Vbuf[2][64 * 32];   // [d][key]       8 KB
  __shared__ _Float16 Pl[4][16 * 32];     // per-wave probs 4 KB

  const int tid  = threadIdx.x;
  const int lane = tid & 31;
  const int wave = tid >> 5;               // 0..3
  const int bh   = blockIdx.y;             // 0..47
  const int b    = bh / kH;
  const int h    = bh % kH;
  const int q0   = blockIdx.x * 64 + wave * 16;

  const _Float16* Qp = Qh + (size_t)bh * kS * kHD;
  const _Float16* Kp = Kh + (size_t)bh * kS * kHD;
  const _Float16* Vp = Vh + (size_t)bh * kHD * kS;  // transposed [64][S]

  const int n     = lane & 15;
  const int mbase = (lane >> 4) << 3;

  // Q fragments, pre-scaled by 1/sqrt(64)=0.125.
  v16h aq0, aq1;
  {
    const _Float16* qrow = Qp + (size_t)(q0 + n) * kHD;
#pragma unroll
    for (int j = 0; j < 16; ++j) {
      const int kk = kmapA(lane, j);
      aq0[j] = qrow[kk]      * (_Float16)0.125f;
      aq1[j] = qrow[32 + kk] * (_Float16)0.125f;
    }
  }

  float qm[8];
#pragma unroll
  for (int j = 0; j < 8; ++j)
    qm[j] = (float)masks[b * kS + q0 + mbase + j];

  float mrun[8], lrun[8];
#pragma unroll
  for (int j = 0; j < 8; ++j) { mrun[j] = -3.0e38f; lrun[j] = 0.0f; }
  v8f acc0 = {}, acc1 = {}, acc2 = {}, acc3 = {};

  // Prologue: wave 0 DMAs tile 0 into buffer 0 via the TDM.
  if (wave == 0) {
    tdm_load_tile_2d((unsigned)(unsigned long long)&Kbuf[0][0], Kp,
                     /*td0*/64, /*td1*/kS, /*tile*/64, 32, /*stride*/64);
    tdm_load_tile_2d((unsigned)(unsigned long long)&Vbuf[0][0], Vp,
                     /*td0*/kS, /*td1*/64, /*tile*/32, 64, /*stride*/kS);
  }

  for (int it = 0; it < kS / 32; ++it) {
    const int kt  = it * 32;
    const int cur = it & 1;
    const _Float16* Kt = &Kbuf[cur][0];
    const _Float16* Vt = &Vbuf[cur][0];

    if (wave == 0) __builtin_amdgcn_s_wait_tensorcnt(0);  // tile `it` landed
    __syncthreads();  // tile visible to all; prev compute done everywhere

    if (wave == 0 && kt + 32 < kS) {  // DMA next tile into the other buffer
      const int nxt = cur ^ 1;
      tdm_load_tile_2d((unsigned)(unsigned long long)&Kbuf[nxt][0],
                       Kp + (size_t)(kt + 32) * kHD, 64, kS, 64, 32, 64);
      tdm_load_tile_2d((unsigned)(unsigned long long)&Vbuf[nxt][0],
                       Vp + (kt + 32), kS, 64, 32, 64, kS);
    }

    // scores for two 16-key sub-tiles
    float sm[2][8];
#pragma unroll
    for (int t = 0; t < 2; ++t) {
      const _Float16* krow = Kt + (size_t)(t * 16 + n) * kHD;
      v16h bk0, bk1;
#pragma unroll
      for (int j = 0; j < 16; ++j) {
        const int kk = kmapB(lane, j);        // two runs of 8 contiguous f16 per lane
        bk0[j] = krow[kk];
        bk1[j] = krow[32 + kk];
      }
      v8f sc = {};
      sc = __builtin_amdgcn_wmma_f32_16x16x32_f16(false, aq0, false, bk0,
                                                  (short)0, sc, false, false);
      sc = __builtin_amdgcn_wmma_f32_16x16x32_f16(false, aq1, false, bk1,
                                                  (short)0, sc, false, false);
      const float km = (float)masks[b * kS + kt + t * 16 + n];
#pragma unroll
      for (int j = 0; j < 8; ++j)
        sm[t][j] = sc[j] + (1.0f - qm[j] * km) * kNegBig;
    }

    // online softmax: per-row reductions within each 16-lane half-wave
#pragma unroll
    for (int j = 0; j < 8; ++j) {
      float mx = fmaxf(sm[0][j], sm[1][j]);
#pragma unroll
      for (int o = 8; o >= 1; o >>= 1) mx = fmaxf(mx, __shfl_xor(mx, o, 32));
      const float mnew = fmaxf(mrun[j], mx);
      const float corr = __expf(mrun[j] - mnew);
      const float p0 = __expf(sm[0][j] - mnew);
      const float p1 = __expf(sm[1][j] - mnew);
      sm[0][j] = p0; sm[1][j] = p1;
      float rs = p0 + p1;
#pragma unroll
      for (int o = 8; o >= 1; o >>= 1) rs += __shfl_xor(rs, o, 32);
      lrun[j] = lrun[j] * corr + rs;
      mrun[j] = mnew;
      acc0[j] *= corr; acc1[j] *= corr; acc2[j] *= corr; acc3[j] *= corr;
    }

    // transpose P (D-fragment) -> A-fragment via per-wave LDS (same-wave DS
    // ops are in-order; no barrier needed)
    _Float16* pw = &Pl[wave][0];
#pragma unroll
    for (int t = 0; t < 2; ++t)
#pragma unroll
      for (int j = 0; j < 8; ++j)
        pw[(mbase + j) * 32 + t * 16 + n] = (_Float16)sm[t][j];

    v16h ap;
    {
      const _Float16* pr = pw + n * 32;
#pragma unroll
      for (int j = 0; j < 16; ++j) ap[j] = pr[kmapA(lane, j)];
    }

    // ctx += P(16x32) x V(32x64); V is [d][key] so each lane reads two
    // 8-f16 contiguous runs per fragment.
    auto pv = [&](int c, v8f acc) -> v8f {
      const _Float16* vrow = Vt + (size_t)(c * 16 + n) * 32;
      v16h bv;
#pragma unroll
      for (int j = 0; j < 16; ++j) bv[j] = vrow[kmapB(lane, j)];
      return __builtin_amdgcn_wmma_f32_16x16x32_f16(false, ap, false, bv,
                                                    (short)0, acc, false,
                                                    false);
    };
    acc0 = pv(0, acc0);
    acc1 = pv(1, acc1);
    acc2 = pv(2, acc2);
    acc3 = pv(3, acc3);

    __syncthreads();  // everyone done with buffer `cur` before it is refilled
  }

  // final normalize + store fp32 [B,S,768]
#pragma unroll
  for (int j = 0; j < 8; ++j) {
    const int q = q0 + mbase + j;
    const float inv = 1.0f / lrun[j];
    float* op = out + ((size_t)(b * kS + q)) * kD + h * kHD;
    op[0 * 16 + n] = acc0[j] * inv;
    op[1 * 16 + n] = acc1[j] * inv;
    op[2 * 16 + n] = acc2[j] * inv;
    op[3 * 16 + n] = acc3[j] * inv;
  }
}

extern "C" void kernel_launch(void* const* d_in, const int* in_sizes, int n_in,
                              void* d_out, int out_size, void* d_ws,
                              size_t ws_size, hipStream_t stream) {
  const float* toks = (const float*)d_in[0];
  const int*  masks = (const int*)d_in[1];
  const float* Wq = (const float*)d_in[2];
  const float* bq = (const float*)d_in[3];
  const float* Wk = (const float*)d_in[4];
  const float* bk = (const float*)d_in[5];
  const float* Wv = (const float*)d_in[6];
  const float* bv = (const float*)d_in[7];
  float* out = (float*)d_out;

  const size_t per = (size_t)kB * kH * kS * kHD;   // 6.29M f16 elements
  _Float16* Qh = (_Float16*)d_ws;
  _Float16* Kh = Qh + per;
  _Float16* Vh = Kh + per;

  dim3 g1(kB * kS / 16, 18);
  qkv_proj_kernel<<<g1, 256, 0, stream>>>(toks, Wq, bq, Wk, bk, Wv, bv,
                                          Qh, Kh, Vh);

  dim3 g2(kS / 64, kB * kH);
  attn_kernel<<<g2, 128, 0, stream>>>(Qh, Kh, Vh, masks, out);
}
